// Block_Attention_85478439125965
// MI455X (gfx1250) — matrix operand order repaired and checked by reading, not compiled, not run on hardware
//
#include <hip/hip_runtime.h>
#include <hip/hip_bf16.h>
#include <math.h>

// ---------------------------------------------------------------------------
// Block attention pipeline for MI455X (gfx1250, wave32, WMMA).
//   x:(4,64,256,256) -> qkv 1x1 conv -> dw3x3 -> resize 256->294 (align=True)
//   -> 7x7-window cosine attention (9216 independent 49x49x49 GEMM pairs,
//      run on v_wmma_f32_16x16x32_f16 with 64x64 zero-padded tiles)
//   -> resize 294->256 (align=False) -> 1x1 proj -> out:(4,64,256,256) f32
// ---------------------------------------------------------------------------

typedef _Float16 v16h __attribute__((ext_vector_type(16)));
typedef _Float16 v8h  __attribute__((ext_vector_type(8)));
typedef float    v8f  __attribute__((ext_vector_type(8)));

#define BATCH 4
#define CIN   64
#define HEADS 8
#define DCH   8
#define IMH   256
#define IMW   256
#define PS    7
#define PS2   49
#define THW   294          // padded spatial (6 * 49)
#define XB    6
#define NWIN  (BATCH * XB * XB)   // 144
#define HW    (IMH * IMW)         // 65536
#define THW2  (THW * THW)         // 86436

// ---------------- Stage 1: 1x1 conv (64 -> 192 channels) -------------------
__global__ void k_conv1x1_qkv(const float* __restrict__ x,
                              const float* __restrict__ w,   // (192,64)
                              float* __restrict__ out) {
  __shared__ float sX[64][32];    // [cin][pixel]
  __shared__ float sW[192][64];   // [cout][cin]
  const int blk = blockIdx.x;
  const int b  = blk >> 11;               // 2048 pixel-tiles per batch
  const int p0 = (blk & 2047) * 32;
  const int tid = threadIdx.x;

  for (int i = tid; i < 64 * 32; i += 256) {
    int ch = i >> 5, px = i & 31;
    sX[ch][px] = x[((size_t)(b * CIN + ch) << 16) + p0 + px];
  }
  for (int i = tid; i < 192 * 64; i += 256) sW[i >> 6][i & 63] = w[i];
  __syncthreads();

  for (int i = tid; i < 192 * 32; i += 256) {
    int o = i >> 5, px = i & 31;
    float s = 0.f;
#pragma unroll
    for (int c = 0; c < 64; ++c) s += sW[o][c] * sX[c][px];
    out[((size_t)(b * 192 + o) << 16) + p0 + px] = s;
  }
}

// ---------------- Stage 2: depthwise 3x3, pad 1 ----------------------------
__global__ void k_dwconv3x3(const float* __restrict__ in,
                            const float* __restrict__ w,   // (192,1,3,3)
                            float* __restrict__ out, int total) {
  int idx = blockIdx.x * blockDim.x + threadIdx.x;
  if (idx >= total) return;
  int xx = idx & 255;
  int yy = (idx >> 8) & 255;
  int bc = idx >> 16;            // 0..767
  int o  = bc % 192;
  const float* src = in + ((size_t)bc << 16);
  float s = 0.f;
#pragma unroll
  for (int ky = 0; ky < 3; ++ky) {
    int y = yy + ky - 1;
    if (y < 0 || y > 255) continue;
#pragma unroll
    for (int kx = 0; kx < 3; ++kx) {
      int x = xx + kx - 1;
      if (x < 0 || x > 255) continue;
      s += w[o * 9 + ky * 3 + kx] * src[(y << 8) + x];
    }
  }
  out[idx] = s;
}

// ---------------- Stage 3: bilinear 256 -> 294, align_corners=True ---------
__global__ void k_resize_up(const float* __restrict__ in, float* __restrict__ out,
                            int total) {
  int idx = blockIdx.x * blockDim.x + threadIdx.x;
  if (idx >= total) return;
  int ox = idx % THW;
  int t  = idx / THW;
  int oy = t % THW;
  int bc = t / THW;              // 0..767
  const float scale = 255.0f / 293.0f;
  float ys = oy * scale, xs = ox * scale;
  int y0 = (int)floorf(ys), x0 = (int)floorf(xs);
  int y1 = min(y0 + 1, 255), x1 = min(x0 + 1, 255);
  float ly = ys - y0, lx = xs - x0;
  const float* s = in + ((size_t)bc << 16);
  float r0 = s[(y0 << 8) + x0] * (1.f - lx) + s[(y0 << 8) + x1] * lx;
  float r1 = s[(y1 << 8) + x0] * (1.f - lx) + s[(y1 << 8) + x1] * lx;
  out[idx] = r0 * (1.f - ly) + r1 * ly;
}

// ---------------- Stage 4: windowed cosine attention (WMMA) ----------------
// window element (i,j) <-> image (r,c):
//   r = xw*49 + (j%7)*7 + (i%7) ; c = yw*49 + (i/7)*7 + (j/7)
__device__ __forceinline__ size_t win_addr(int bidx, int nch, int ch, int xw,
                                           int yw, int i, int j) {
  int r = xw * PS2 + (j % 7) * 7 + (i % 7);
  int c = yw * PS2 + (i / 7) * 7 + (j / 7);
  return ((size_t)(bidx * nch + ch) * THW + r) * THW + c;
}

__device__ __forceinline__ v16h cat8(v8h lo, v8h hi) {
  v16h r;
#pragma unroll
  for (int t = 0; t < 8; ++t) { r[t] = lo[t]; r[t + 8] = hi[t]; }
  return r;
}
// A fragment (16x32 f16): lane l<16 -> row M0+l, halves K0+[0..7] & K0+16+[0..7]
//                          lane>=16 -> row M0+l, halves K0+8+[0..7] & K0+24+[0..7]
__device__ __forceinline__ v16h frag_a(const _Float16* base, int M0, int K0,
                                       int hl, int l) {
  const _Float16* r = base + (M0 + l) * 72;
  v8h lo = *(const v8h*)(r + K0 + hl * 8);
  v8h hi = *(const v8h*)(r + K0 + 16 + hl * 8);
  return cat8(lo, hi);
}
// B fragment (32x16 f16): lane l<16 -> col N0+l, halves K0+[0..7] & K0+16+[0..7]
//                          lane>=16 -> col N0+l, halves K0+8+[0..7] & K0+24+[0..7]
// (matrix stored so that "row index == N, col index == K" is contiguous)
__device__ __forceinline__ v16h frag_b(const _Float16* base, int N0, int K0,
                                       int hl, int l) {
  const _Float16* r = base + (N0 + l) * 72;
  v8h lo = *(const v8h*)(r + K0 + hl * 8);
  v8h hi = *(const v8h*)(r + K0 + 16 + hl * 8);
  return cat8(lo, hi);
}

__global__ void __launch_bounds__(128)
k_win_attention(const float* __restrict__ qkv,   // (4,192,294,294)
                float* __restrict__ outimg) {    // (4, 64,294,294)
  __shared__ __attribute__((aligned(16))) _Float16 sQ[64][72];
  __shared__ __attribute__((aligned(16))) _Float16 sK[64][72];
  __shared__ __attribute__((aligned(16))) _Float16 sVT[64][72];  // v transposed
  __shared__ __attribute__((aligned(16))) _Float16 sP[64][72];
  __shared__ float sS[64][65];

  const int bid  = blockIdx.x;          // 9216 = 144 windows * 64 (head,d)
  const int hd   = bid & 63;
  const int wid  = bid >> 6;
  const int head = hd >> 3, dc = hd & 7;
  const int bidx = wid / 36;
  const int rem  = wid % 36;
  const int xw   = rem / 6, yw = rem % 6;
  const int ch   = head * DCH + dc;     // 0..63 within q/k/v block

  const int tid = threadIdx.x;

  // ---- load + L2-normalize q (threads 0..63) and k (threads 64..127) ----
  {
    int row = (tid < 64) ? tid : tid - 64;
    _Float16(*dst)[72] = (tid < 64) ? sQ : sK;
    int gch = (tid < 64) ? ch : (64 + ch);
    if (row < PS2) {
      float vals[PS2];
      float ss = 0.f;
#pragma unroll
      for (int j = 0; j < PS2; ++j) {
        float v = qkv[win_addr(bidx, 192, gch, xw, yw, row, j)];
        vals[j] = v;
        ss += v * v;
      }
      float inv = 1.0f / fmaxf(sqrtf(ss), 1e-12f);
#pragma unroll
      for (int j = 0; j < PS2; ++j) dst[row][j] = (_Float16)(vals[j] * inv);
#pragma unroll
      for (int j = PS2; j < 64; ++j) dst[row][j] = (_Float16)0.f;
    } else {
#pragma unroll
      for (int j = 0; j < 64; ++j) dst[row][j] = (_Float16)0.f;
    }
  }
  // ---- load v transposed: sVT[n][k] = v[k][n] (all 128 threads) ----
  {
    int n  = tid >> 1;
    int c0 = (tid & 1) * 32;
#pragma unroll
    for (int kk = c0; kk < c0 + 32; ++kk) {
      _Float16 v = (_Float16)0.f;
      if (kk < PS2 && n < PS2)
        v = (_Float16)qkv[win_addr(bidx, 192, 128 + ch, xw, yw, kk, n)];
      sVT[n][kk] = v;
    }
  }
  __syncthreads();

  const int wave = tid >> 5;
  const int lane = tid & 31;
  const int hl = lane >> 4, l = lane & 15;
  const int M0 = wave * 16;

  // ---- S = qn @ kn^T  (64x64 padded, K = 64) ----
#pragma unroll
  for (int nt = 0; nt < 4; ++nt) {
    const int N0 = nt * 16;
    v8f acc = {0.f, 0.f, 0.f, 0.f, 0.f, 0.f, 0.f, 0.f};
#pragma unroll
    for (int kc = 0; kc < 2; ++kc) {
      const int K0 = kc * 32;
      v16h a = frag_a(&sQ[0][0], M0, K0, hl, l);
      v16h b = frag_b(&sK[0][0], N0, K0, hl, l);
      acc = __builtin_amdgcn_wmma_f32_16x16x32_f16(false, a, false, b,
                                                   (short)0, acc, false, false);
    }
#pragma unroll
    for (int r = 0; r < 8; ++r) sS[M0 + hl * 8 + r][N0 + l] = acc[r];
  }
  __syncthreads();

  // ---- row softmax over j = 0..48; write P (f16, zero padded) ----
  if (tid < 64) {
    int row = tid;
    if (row < PS2) {
      float mx = -3.4e38f;
#pragma unroll
      for (int j = 0; j < PS2; ++j) mx = fmaxf(mx, sS[row][j]);
      float e[PS2];
      float sum = 0.f;
#pragma unroll
      for (int j = 0; j < PS2; ++j) { e[j] = __expf(sS[row][j] - mx); sum += e[j]; }
      float inv = 1.0f / sum;
#pragma unroll
      for (int j = 0; j < PS2; ++j) sP[row][j] = (_Float16)(e[j] * inv);
#pragma unroll
      for (int j = PS2; j < 64; ++j) sP[row][j] = (_Float16)0.f;
    } else {
#pragma unroll
      for (int j = 0; j < 64; ++j) sP[row][j] = (_Float16)0.f;
    }
  }
  __syncthreads();

  // ---- O = P @ V ; scatter window (i,j) back to padded image ----
  const size_t obase = (size_t)(bidx * 64 + ch) * THW2;
#pragma unroll
  for (int nt = 0; nt < 4; ++nt) {
    const int N0 = nt * 16;
    v8f acc = {0.f, 0.f, 0.f, 0.f, 0.f, 0.f, 0.f, 0.f};
#pragma unroll
    for (int kc = 0; kc < 2; ++kc) {
      const int K0 = kc * 32;
      v16h a = frag_a(&sP[0][0], M0, K0, hl, l);
      v16h b = frag_b(&sVT[0][0], N0, K0, hl, l);
      acc = __builtin_amdgcn_wmma_f32_16x16x32_f16(false, a, false, b,
                                                   (short)0, acc, false, false);
    }
#pragma unroll
    for (int r = 0; r < 8; ++r) {
      int i = M0 + hl * 8 + r;
      int j = N0 + l;
      if (i < PS2 && j < PS2) {
        int rr = xw * PS2 + (j % 7) * 7 + (i % 7);
        int cc = yw * PS2 + (i / 7) * 7 + (j / 7);
        outimg[obase + (size_t)rr * THW + cc] = acc[r];
      }
    }
  }
}

// ---------------- Stage 5: bilinear 294 -> 256, align_corners=False --------
__global__ void k_resize_down(const float* __restrict__ in, float* __restrict__ out,
                              int total) {
  int idx = blockIdx.x * blockDim.x + threadIdx.x;
  if (idx >= total) return;
  int ox = idx & 255;
  int oy = (idx >> 8) & 255;
  int bc = idx >> 16;            // 0..255
  const float scale = (float)THW / 256.0f;
  float ys = fmaxf((oy + 0.5f) * scale - 0.5f, 0.0f);
  float xs = fmaxf((ox + 0.5f) * scale - 0.5f, 0.0f);
  int y0 = (int)ys, x0 = (int)xs;
  int y1 = min(y0 + 1, THW - 1), x1 = min(x0 + 1, THW - 1);
  float ly = ys - y0, lx = xs - x0;
  const float* s = in + (size_t)bc * THW2;
  float r0 = s[y0 * THW + x0] * (1.f - lx) + s[y0 * THW + x1] * lx;
  float r1 = s[y1 * THW + x0] * (1.f - lx) + s[y1 * THW + x1] * lx;
  out[idx] = r0 * (1.f - ly) + r1 * ly;
}

// ---------------- Stage 6: 1x1 projection (64 -> 64) -----------------------
__global__ void k_proj(const float* __restrict__ in,
                       const float* __restrict__ w,   // (64,64)
                       float* __restrict__ out) {
  __shared__ float sX[64][64];
  __shared__ float sW[64][64];
  const int blk = blockIdx.x;
  const int b  = blk >> 10;
  const int p0 = (blk & 1023) * 64;
  const int tid = threadIdx.x;
  for (int i = tid; i < 64 * 64; i += 256) {
    int ch = i >> 6, px = i & 63;
    sX[ch][px] = in[((size_t)(b * 64 + ch) << 16) + p0 + px];
    sW[ch][px] = w[i];           // sW[o][c]
  }
  __syncthreads();
  for (int i = tid; i < 64 * 64; i += 256) {
    int o = i >> 6, px = i & 63;
    float s = 0.f;
#pragma unroll
    for (int c = 0; c < 64; ++c) s += sW[o][c] * sX[c][px];
    out[((size_t)(b * 64 + o) << 16) + p0 + px] = s;
  }
}

// ---------------------------------------------------------------------------
extern "C" void kernel_launch(void* const* d_in, const int* in_sizes, int n_in,
                              void* d_out, int out_size, void* d_ws, size_t ws_size,
                              hipStream_t stream) {
  (void)in_sizes; (void)n_in; (void)out_size; (void)ws_size;
  const float* x      = (const float*)d_in[0];
  const float* qkv_w  = (const float*)d_in[1];
  const float* dw_w   = (const float*)d_in[2];
  const float* proj_w = (const float*)d_in[3];
  float* out = (float*)d_out;
  float* ws  = (float*)d_ws;

  // Workspace layout (floats), with liveness-based aliasing:
  //   P0: conv1x1 out (4*192*65536)  then resized qkv (4*192*86436)
  //   P1: dwconv out  (4*192*65536)  then attention out image (4*64*86436)
  //   P2: downsampled attention image (4*64*65536)
  const size_t P0 = 0;
  const size_t P1 = (size_t)4 * 192 * THW2;                 // 66,382,848
  const size_t P2 = P1 + (size_t)4 * 192 * HW;              // + 50,331,648
  float* bufA = ws + P0;   // conv1x1 out
  float* bufC = ws + P0;   // resized qkv (aliases A after A is dead)
  float* bufB = ws + P1;   // dwconv out
  float* bufD = ws + P1;   // attention out image (aliases B after B is dead)
  float* bufE = ws + P2;   // 294->256 resized attention output

  const int n_dw = 4 * 192 * HW;       // 50,331,648
  const int n_up = 4 * 192 * THW2;     // 66,382,848
  const int n_dn = 4 * 64 * HW;        // 16,777,216

  k_conv1x1_qkv<<<4 * 2048, 256, 0, stream>>>(x, qkv_w, bufA);
  k_dwconv3x3 <<<(n_dw + 255) / 256, 256, 0, stream>>>(bufA, dw_w, bufB, n_dw);
  k_resize_up <<<(n_up + 255) / 256, 256, 0, stream>>>(bufB, bufC, n_up);
  k_win_attention<<<NWIN * HEADS * DCH, 128, 0, stream>>>(bufC, bufD);
  k_resize_down<<<(n_dn + 255) / 256, 256, 0, stream>>>(bufD, bufE, n_dn);
  k_proj      <<<4 * 1024, 256, 0, stream>>>(bufE, proj_w, out);
}
